// SelfAttention_24627342475460
// MI455X (gfx1250) — matrix operand order repaired and checked by reading, not compiled, not run on hardware
//
#include <hip/hip_runtime.h>
#include <hip/hip_bf16.h>
#include <math.h>

// ---------------------------------------------------------------------------
// Self-attention block (GN -> QKV -> softmax(QK^T/sqrt(C))V -> proj -> +x)
// b=4, C=256, H=W=64, N=4096.  All GEMMs via v_wmma_f32_16x16x32_f16.
// Flash-attention style: scores never touch HBM (~1GB fp32 avoided).
// K/V tiles double-buffered into LDS with global_load_async_to_lds_b128
// (ASYNCcnt-tracked), shared by all waves of the block.
// ---------------------------------------------------------------------------

typedef _Float16 f16;
typedef __attribute__((ext_vector_type(16))) _Float16 v16h;
typedef __attribute__((ext_vector_type(8)))  _Float16 v8h;
typedef __attribute__((ext_vector_type(8)))  float    v8f;

#define BATCH 4
#define CCH   256
#define NPIX  4096
#define QS    264   // q-tile LDS row stride (halves), 256 -> 264 pad
#define KS    264   // k-tile LDS row stride (halves)
#define VS    40    // v-tile LDS row stride (halves), 32 -> 40 pad
#define PS    40    // P-tile LDS row stride (halves)

static __device__ __forceinline__ v16h cat16(v8h lo, v8h hi) {
  v16h r;
#pragma unroll
  for (int i = 0; i < 8; ++i) { r[i] = lo[i]; r[i + 8] = hi[i]; }
  return r;
}

static __device__ __forceinline__ v8f wmma16(v16h a, v16h b, v8f c) {
  // D = A(16x32 f16) * B(32x16 f16) + C(16x16 f32)
  return __builtin_amdgcn_wmma_f32_16x16x32_f16(false, a, false, b, (short)0, c,
                                                false, false);
}

// Async 16B copy global -> LDS (per-lane), tracked by ASYNCcnt.
// VDST carries the per-lane LDS byte offset (added to the LDS base and the instruction offset).
static __device__ __forceinline__ void async_b128(uint32_t lds_addr,
                                                  const void* gptr) {
  asm volatile("global_load_async_to_lds_b128 %0, %1, off"
               :: "v"(lds_addr), "v"((uint64_t)(size_t)gptr)
               : "memory");
}

// ---------------- GroupNorm statistics (deterministic 2-stage) -------------
__global__ void gn_partial_kernel(const float* __restrict__ x,
                                  float* __restrict__ part) {
  const int b = blockIdx.y, blk = blockIdx.x;
  const int SLAB = (CCH * NPIX) / 64;          // 16384 elems per block
  const size_t base = (size_t)b * CCH * NPIX + (size_t)blk * SLAB;
  float s = 0.f, s2 = 0.f;
  for (int i = threadIdx.x; i < SLAB; i += 256) {
    float v = x[base + i];
    s += v; s2 += v * v;
  }
  __shared__ float r0[256], r1[256];
  r0[threadIdx.x] = s; r1[threadIdx.x] = s2;
  __syncthreads();
  for (int off = 128; off > 0; off >>= 1) {
    if ((int)threadIdx.x < off) {
      r0[threadIdx.x] += r0[threadIdx.x + off];
      r1[threadIdx.x] += r1[threadIdx.x + off];
    }
    __syncthreads();
  }
  if (threadIdx.x == 0) {
    part[(b * 64 + blk) * 2 + 0] = r0[0];
    part[(b * 64 + blk) * 2 + 1] = r1[0];
  }
}

__global__ void gn_final_kernel(const float* __restrict__ part,
                                float* __restrict__ stats) {
  const int b = blockIdx.x;
  __shared__ float r0[64], r1[64];
  r0[threadIdx.x] = part[(b * 64 + threadIdx.x) * 2 + 0];
  r1[threadIdx.x] = part[(b * 64 + threadIdx.x) * 2 + 1];
  __syncthreads();
  for (int off = 32; off > 0; off >>= 1) {
    if ((int)threadIdx.x < off) {
      r0[threadIdx.x] += r0[threadIdx.x + off];
      r1[threadIdx.x] += r1[threadIdx.x + off];
    }
    __syncthreads();
  }
  if (threadIdx.x == 0) {
    const float M = (float)(CCH * NPIX);
    float mu  = r0[0] / M;
    float var = r1[0] / M - mu * mu;
    stats[b * 2 + 0] = mu;
    stats[b * 2 + 1] = rsqrtf(var + 1e-6f);
  }
}

// ---------------- Weight f32 -> f16 conversion -----------------------------
__global__ void wconv_kernel(const float* __restrict__ wq,
                             const float* __restrict__ wk,
                             const float* __restrict__ wv,
                             const float* __restrict__ wo,
                             f16* __restrict__ dst) {
  int idx = blockIdx.x * 256 + threadIdx.x;   // 0 .. 4*65536-1
  int which = idx >> 16, off = idx & 65535;
  const float* src = (which == 0) ? wq : (which == 1) ? wk : (which == 2) ? wv : wo;
  dst[idx] = (f16)src[off];
}

// ---------------- QKV projection (GN fused), WMMA GEMMs --------------------
// Block: 128 threads (4 waves), 64 pixel rows per block. Each wave: 16 rows.
// q,k stored (b,N,C) f16 (pixel-major); v stored (b,C,N) f16 (channel-major).
__global__ __launch_bounds__(128, 1) void qkv_kernel(
    const float* __restrict__ x, const float* __restrict__ gnw,
    const float* __restrict__ gnb, const f16* __restrict__ wqh,
    const f16* __restrict__ wkh, const f16* __restrict__ wvh,
    const float* __restrict__ bq, const float* __restrict__ bk,
    const float* __restrict__ bv, const float* __restrict__ stats,
    f16* __restrict__ qh, f16* __restrict__ kh, f16* __restrict__ vh) {
  __shared__ f16 hlds[64 * QS];                     // normalized tile, ~33 KB
  const int b = blockIdx.y;
  const int i0 = blockIdx.x * 64;
  const float mu = stats[b * 2 + 0];
  const float rinv = stats[b * 2 + 1];

  // Stage: read x (c-major) coalesced, write normalized f16 pixel-major.
  for (int idx = threadIdx.x; idx < 64 * CCH; idx += 128) {
    int c = idx >> 6, il = idx & 63;
    float v = x[((size_t)b * CCH + c) * NPIX + i0 + il];
    float hn = (v - mu) * rinv * gnw[c] + gnb[c];
    hlds[il * QS + c] = (f16)hn;
  }
  __syncthreads();

  const int lane = threadIdx.x & 31, wave = threadIdx.x >> 5;
  const int g = lane >> 4, lh = lane & 15;
  const int ilb = wave * 16;
  const float qscale = 0.0625f;                     // C^-0.5 = 1/16

  f16* qb = qh + (size_t)b * NPIX * CCH;
  f16* kb = kh + (size_t)b * NPIX * CCH;
  f16* vb = vh + (size_t)b * CCH * NPIX;

#pragma unroll 1
  for (int cc = 0; cc < 16; ++cc) {
    v8f aq = {}, ak = {}, av = {};
    const int o = cc * 16 + lh;                     // output channel (D column)
#pragma unroll
    for (int kc = 0; kc < 8; ++kc) {
      const f16* pa = &hlds[(ilb + lh) * QS + kc * 32 + 8 * g];
      v16h A = cat16(*(const v8h*)pa, *(const v8h*)(pa + 16));
      const f16* pq = &wqh[(size_t)o * CCH + kc * 32 + 8 * g];
      const f16* pk = &wkh[(size_t)o * CCH + kc * 32 + 8 * g];
      const f16* pv = &wvh[(size_t)o * CCH + kc * 32 + 8 * g];
      v16h Bq = cat16(*(const v8h*)pq, *(const v8h*)(pq + 16));
      v16h Bk = cat16(*(const v8h*)pk, *(const v8h*)(pk + 16));
      v16h Bv = cat16(*(const v8h*)pv, *(const v8h*)(pv + 16));
      aq = wmma16(A, Bq, aq);
      ak = wmma16(A, Bk, ak);
      av = wmma16(A, Bv, av);
    }
    const float biasq = bq[o], biask = bk[o], biasv = bv[o];
#pragma unroll
    for (int r = 0; r < 8; ++r) {
      const int i = i0 + ilb + r + 8 * g;           // D row = vgpr + 8*lanegroup
      qb[(size_t)i * CCH + o] = (f16)((aq[r] + biasq) * qscale);
      kb[(size_t)i * CCH + o] = (f16)(ak[r] + biask);
      vb[(size_t)o * NPIX + i] = (f16)(av[r] + biasv);
    }
  }
}

// ---------------- Flash attention ------------------------------------------
// Block: 128 threads (4 waves), 64 query rows; wave owns 16 rows.
// Per iteration: 32 keys. K tile (32x256 f16) and V tile (256x32 f16) are
// shared by all waves -> staged into LDS with async b128 copies, double
// buffered; waves overlap the next tile's fill with this tile's WMMAs.
// Dynamic LDS layout (halves):
//   [0)            q tile      64 x QS
//   [qoff)         k tiles     2 x 32 x KS
//   [voff)         v tiles     2 x 256 x VS
//   [poff)         P bounce    4 x 16 x PS
__global__ __launch_bounds__(128, 1) void attn_kernel(
    const f16* __restrict__ qh, const f16* __restrict__ kh,
    const f16* __restrict__ vh, f16* __restrict__ oh) {
  extern __shared__ f16 smem[];
  f16* qlds = smem;                                 // 16896 halves
  f16* ktl  = smem + 64 * QS;                       // 2 x 8448
  f16* vtl  = ktl + 2 * 32 * KS;                    // 2 x 10240
  f16* plds = vtl + 2 * 256 * VS;                   // 4 x 640

  const int b = blockIdx.y;
  const int i0 = blockIdx.x * 64;
  const f16* qg = qh + (size_t)b * NPIX * CCH;
  const f16* kg = kh + (size_t)b * NPIX * CCH;
  const f16* vg = vh + (size_t)b * CCH * NPIX;

  // Stage q tile (read-only for the whole kernel).
  for (int idx = threadIdx.x; idx < (64 * CCH) / 8; idx += 128) {
    int il = idx >> 5, coff = (idx & 31) * 8;
    *(v8h*)&qlds[il * QS + coff] = *(const v8h*)&qg[(size_t)(i0 + il) * CCH + coff];
  }

  // Async stage of one 32-key K/V tile pair into buffer `buf`.
  // 16 b128 ops per thread per fill (8 for K, 8 for V).
  auto stage_tile = [&](int buf, int jt) {
#pragma unroll 2
    for (int ch = threadIdx.x; ch < 1024; ch += 128) {      // K: 32 x 512B
      int row = ch >> 5, c16 = ch & 31;
      async_b128((uint32_t)(size_t)&ktl[buf * 32 * KS + row * KS + c16 * 8],
                 &kg[(size_t)(jt + row) * CCH + c16 * 8]);
    }
#pragma unroll 2
    for (int ch = threadIdx.x; ch < 1024; ch += 128) {      // V: 256 x 64B
      int c = ch >> 2, j8 = ch & 3;
      async_b128((uint32_t)(size_t)&vtl[buf * 256 * VS + c * VS + j8 * 8],
                 &vg[(size_t)c * NPIX + jt + j8 * 8]);
    }
  };

  stage_tile(0, 0);

  const int lane = threadIdx.x & 31, wave = threadIdx.x >> 5;
  const int g = lane >> 4, lh = lane & 15;
  const int ilb = wave * 16;
  f16* pw = &plds[wave * 16 * PS];

  v8f O[16];
#pragma unroll
  for (int cc = 0; cc < 16; ++cc) { v8f z = {}; O[cc] = z; }
  float m[8], l[8];
#pragma unroll
  for (int r = 0; r < 8; ++r) { m[r] = -3.0e38f; l[r] = 0.f; }

  __syncthreads();   // q tile visible

  for (int jt = 0; jt < NPIX; jt += 32) {
    const int buf = (jt >> 5) & 1;
    // Prefetch next tile into the other buffer (wraps on last iter; unused).
    stage_tile(buf ^ 1, (jt + 32) & (NPIX - 1));
    // 32 async ops outstanding; wait for the oldest 16 (= current buffer).
    asm volatile("s_wait_asynccnt 0x10" ::: "memory");
    __syncthreads();                                // all waves' fills visible

    const f16* kbuf = &ktl[buf * 32 * KS];
    const f16* vbuf = &vtl[buf * 256 * VS];

    // ---- scores: S0 (keys jt..jt+15), S1 (keys jt+16..jt+31), K=256
    v8f S0 = {}, S1 = {};
#pragma unroll
    for (int kc = 0; kc < 8; ++kc) {
      const f16* pa = &qlds[(ilb + lh) * QS + kc * 32 + 8 * g];
      v16h A = cat16(*(const v8h*)pa, *(const v8h*)(pa + 16));
      const f16* pb0 = &kbuf[lh * KS + kc * 32 + 8 * g];
      const f16* pb1 = &kbuf[(16 + lh) * KS + kc * 32 + 8 * g];
      v16h B0 = cat16(*(const v8h*)pb0, *(const v8h*)(pb0 + 16));
      v16h B1 = cat16(*(const v8h*)pb1, *(const v8h*)(pb1 + 16));
      S0 = wmma16(A, B0, S0);
      S1 = wmma16(A, B1, S1);
    }

    // ---- online softmax per row (row = r + 8*lanegroup, cols across lanes)
#pragma unroll
    for (int r = 0; r < 8; ++r) {
      float s0 = S0[r], s1 = S1[r];
      float mx = fmaxf(s0, s1);
      mx = fmaxf(mx, __shfl_xor(mx, 1));
      mx = fmaxf(mx, __shfl_xor(mx, 2));
      mx = fmaxf(mx, __shfl_xor(mx, 4));
      mx = fmaxf(mx, __shfl_xor(mx, 8));            // row max (16-lane half)
      float mnew = fmaxf(m[r], mx);
      float alpha = __expf(m[r] - mnew);
      float p0 = __expf(s0 - mnew);
      float p1 = __expf(s1 - mnew);
      float rs = p0 + p1;
      rs += __shfl_xor(rs, 1);
      rs += __shfl_xor(rs, 2);
      rs += __shfl_xor(rs, 4);
      rs += __shfl_xor(rs, 8);                      // row sum
      l[r] = l[r] * alpha + rs;
      m[r] = mnew;
#pragma unroll
      for (int cc = 0; cc < 16; ++cc) O[cc][r] *= alpha;   // rescale running O
      // D layout -> LDS (row-major P tile) for A-layout reload
      pw[(r + 8 * g) * PS + lh] = (f16)p0;
      pw[(r + 8 * g) * PS + 16 + lh] = (f16)p1;
    }
    asm volatile("s_wait_dscnt 0x0" ::: "memory");  // wave-private LDS RAW

    // ---- reload P as A matrix (16 rows x 32 keys)
    const f16* pa = &pw[lh * PS + 8 * g];
    v16h P = cat16(*(const v8h*)pa, *(const v8h*)(pa + 16));

    // ---- O += P * V^T  (V tile channel-major: contiguous along keys)
#pragma unroll
    for (int cc = 0; cc < 16; ++cc) {
      const f16* pv = &vbuf[(cc * 16 + lh) * VS + 8 * g];
      v16h Bv = cat16(*(const v8h*)pv, *(const v8h*)(pv + 16));
      O[cc] = wmma16(P, Bv, O[cc]);
    }
    __syncthreads();   // everyone done reading `buf` before it is re-filled
  }

  // ---- normalize and store O as f16 (b,N,C)
  f16* ob = oh + (size_t)b * NPIX * CCH;
  float linv[8];
#pragma unroll
  for (int r = 0; r < 8; ++r) linv[r] = 1.0f / l[r];
#pragma unroll
  for (int cc = 0; cc < 16; ++cc) {
    const int c = cc * 16 + lh;
#pragma unroll
    for (int r = 0; r < 8; ++r) {
      const int i = i0 + ilb + r + 8 * g;
      ob[(size_t)i * CCH + c] = (f16)(O[cc][r] * linv[r]);
    }
  }
}

// ---------------- Output projection + bias + residual ----------------------
__global__ __launch_bounds__(256, 1) void proj_kernel(
    const f16* __restrict__ oh, const f16* __restrict__ woh,
    const float* __restrict__ bo, const float* __restrict__ x,
    float* __restrict__ out) {
  const int b = blockIdx.y;
  const int i0 = blockIdx.x * 128;
  const int lane = threadIdx.x & 31, wave = threadIdx.x >> 5;
  const int g = lane >> 4, lh = lane & 15;
  const int ilb = wave * 16;
  const f16* ob = oh + (size_t)b * NPIX * CCH;

#pragma unroll 1
  for (int cc = 0; cc < 16; ++cc) {
    v8f acc = {};
    const int o = cc * 16 + lh;
#pragma unroll
    for (int kc = 0; kc < 8; ++kc) {
      const f16* pa = &ob[(size_t)(i0 + ilb + lh) * CCH + kc * 32 + 8 * g];
      v16h A = cat16(*(const v8h*)pa, *(const v8h*)(pa + 16));
      const f16* pb = &woh[(size_t)o * CCH + kc * 32 + 8 * g];
      v16h B = cat16(*(const v8h*)pb, *(const v8h*)(pb + 16));
      acc = wmma16(A, B, acc);
    }
    const float bias = bo[o];
#pragma unroll
    for (int r = 0; r < 8; ++r) {
      const int i = i0 + ilb + r + 8 * g;
      const size_t adr = ((size_t)b * CCH + o) * NPIX + i;
      out[adr] = x[adr] + acc[r] + bias;
    }
  }
}

// ---------------------------------------------------------------------------
extern "C" void kernel_launch(void* const* d_in, const int* in_sizes, int n_in,
                              void* d_out, int out_size, void* d_ws,
                              size_t ws_size, hipStream_t stream) {
  const float* x   = (const float*)d_in[0];
  const float* gnw = (const float*)d_in[1];
  const float* gnb = (const float*)d_in[2];
  const float* wq  = (const float*)d_in[3];
  const float* bq  = (const float*)d_in[4];
  const float* wk  = (const float*)d_in[5];
  const float* bk  = (const float*)d_in[6];
  const float* wv  = (const float*)d_in[7];
  const float* bv  = (const float*)d_in[8];
  const float* wo  = (const float*)d_in[9];
  const float* bo  = (const float*)d_in[10];
  float* out = (float*)d_out;

  // Workspace layout (requires ~33.5 MB):
  char* ws = (char*)d_ws;
  float* part  = (float*)(ws + 0);            // 4*64*2 f32
  float* stats = (float*)(ws + 4096);         // 4*2   f32
  f16* wqh = (f16*)(ws + 8192);               // 4 x 64K halves (q,k,v,o)
  f16* wkh = wqh + 65536;
  f16* wvh = wkh + 65536;
  f16* woh = wvh + 65536;
  const size_t PLANE = (size_t)BATCH * NPIX * CCH;  // 4M halves = 8 MB
  f16* qh = (f16*)(ws + (1u << 20));
  f16* kh = qh + PLANE;
  f16* vh = kh + PLANE;
  f16* oh = vh + PLANE;

  gn_partial_kernel<<<dim3(64, BATCH), 256, 0, stream>>>(x, part);
  gn_final_kernel<<<BATCH, 64, 0, stream>>>(part, stats);
  wconv_kernel<<<(4 * 65536) / 256, 256, 0, stream>>>(wq, wk, wv, wo, wqh);
  qkv_kernel<<<dim3(NPIX / 64, BATCH), 128, 0, stream>>>(
      x, gnw, gnb, wqh, wkh, wvh, bq, bk, bv, stats, qh, kh, vh);

  // attn dynamic LDS: q(64*QS) + 2*K(32*KS) + 2*V(256*VS) + P(4*16*PS) halves
  const size_t attn_lds =
      (size_t)(64 * QS + 2 * 32 * KS + 2 * 256 * VS + 4 * 16 * PS) * sizeof(f16);
  attn_kernel<<<dim3(NPIX / 64, BATCH), 128, attn_lds, stream>>>(qh, kh, vh, oh);

  proj_kernel<<<dim3(NPIX / 128, BATCH), 256, 0, stream>>>(oh, woh, bo, x, out);
}
